// CrossAttentionBlock_38663295598639
// MI455X (gfx1250) — compile-verified, matrix operand order checked by reading, not run on hardware
//
#include <hip/hip_runtime.h>
#include <hip/hip_bf16.h>
#include <math.h>

// Problem constants (from reference): B=4, C_in=C_out=256, H=W=64 -> N=4096
#define C_DIM 256
#define N_DIM 4096
#define B_DIM 4

// Workspace layout (bf16): Q[b][n][c] | Kt[b][n][c] | V[b][c][n]
// total = 3 * 4 * 4096 * 256 * 2 bytes = 24 MB  (requires ws_size >= 24 MB)

typedef __attribute__((ext_vector_type(16))) __bf16   v16bf;
typedef __attribute__((ext_vector_type(8)))  float    v8f;
typedef __attribute__((ext_vector_type(4)))  unsigned int u32x4;

union Frag {
    v16bf        v;
    unsigned int u[8];
    u32x4        q[2];
};

__device__ __forceinline__ unsigned short f2bf(float f) {
    union { float f; unsigned int u; } cv; cv.f = f;
    unsigned int u = cv.u;
    u += 0x7FFFu + ((u >> 16) & 1u);   // round-to-nearest-even
    return (unsigned short)(u >> 16);
}

__device__ __forceinline__ v8f wmma_bf16(const Frag& a, const Frag& b, v8f c) {
    // (neg_a, A, neg_b, B, c_mod, C, reuse_a, reuse_b)
    return __builtin_amdgcn_wmma_f32_16x16x32_bf16(false, a.v, false, b.v,
                                                   (short)0, c, false, false);
}

// ---------------------------------------------------------------------------
// Kernel 1: the three 1x1-conv projections as bf16 WMMA GEMMs.
// MODE 0: out[n][o] = sum_c in[c][n] * w[o][c]   (Q and Kt, layout [n][c])
// MODE 1: out[o][n] = sum_c w[o][c] * in[c][n]   (V, layout [c][n])
// One wave computes one 16x16 output tile, K = 256 via 8 WMMAs.
// MODE is a template parameter -> no branches in the fragment-build loop.
// ---------------------------------------------------------------------------
template <int MODE>
__global__ __launch_bounds__(128)
void proj_kernel(const float* __restrict__ inA, const float* __restrict__ wA,
                 unsigned short* __restrict__ outA,
                 const float* __restrict__ inB, const float* __restrict__ wB,
                 unsigned short* __restrict__ outB)
{
    const int lane = threadIdx.x & 31;
    const int wid  = threadIdx.x >> 5;
    const int ml   = lane & 15;      // A-row / B-col / D-col within tile
    const int hl   = lane >> 4;      // lane half (K-split per fragment layout)
    const int bb   = blockIdx.z;
    const size_t NC = (size_t)N_DIM * C_DIM;

    const float* in; const float* w; unsigned short* out;
    if (blockIdx.y == 0) { in = inA; w = wA; out = outA; }
    else                 { in = inB; w = wB; out = outB; }
    in  += (size_t)bb * NC;
    out += (size_t)bb * NC;

    const int t  = blockIdx.x * 4 + wid;   // 0..4095 tiles
    const int n0 = (t >> 4) * 16;
    const int o0 = (t & 15) * 16;

    // per-lane base pointers (keep inner-loop address math to one add)
    const float* act_base = in + n0 + ml;                    // strided by N over c
    const float* w_base   = w + (size_t)(o0 + ml) * C_DIM;   // contiguous over c

    v8f acc = {};
    for (int c0 = 0; c0 < C_DIM; c0 += 32) {
        Frag A, Bf;
        #pragma unroll
        for (int v = 0; v < 8; ++v) {
            const int k0 = ((v >> 2) << 4) + hl * 8 + ((v & 3) << 1);
            const int c  = c0 + k0;
            const unsigned int a0 = f2bf(act_base[(size_t)c * N_DIM]);
            const unsigned int a1 = f2bf(act_base[(size_t)(c + 1) * N_DIM]);
            const unsigned int w0 = f2bf(w_base[c]);
            const unsigned int w1 = f2bf(w_base[c + 1]);
            if (MODE == 0) { A.u[v] = a0 | (a1 << 16); Bf.u[v] = w0 | (w1 << 16); }
            else           { A.u[v] = w0 | (w1 << 16); Bf.u[v] = a0 | (a1 << 16); }
        }
        acc = wmma_bf16(A, Bf, acc);
    }

    #pragma unroll
    for (int r = 0; r < 8; ++r) {
        const int row = r + 8 * hl;
        if (MODE == 0)
            out[(size_t)(n0 + row) * C_DIM + o0 + ml] = f2bf(acc[r]);
        else
            out[(size_t)(o0 + row) * N_DIM + n0 + ml] = f2bf(acc[r]);
    }
}

// ---------------------------------------------------------------------------
// Kernel 2: flash attention. One wave owns 16 query rows and all 256 output
// channels (16 f32 accum fragments). Loops over 32 key columns per step:
//   S = Q*K^T (16 WMMAs), online softmax (shfl reductions within 16-lane
//   halves), P -> LDS -> A-fragment, O += P*V (16 WMMAs).
// O rescale (128 v_mul) is skipped wave-uniformly when no row max moved.
// Epilogue: out = gamma * O / l + x.
// ---------------------------------------------------------------------------
__global__ __launch_bounds__(128)
void attn_kernel(const float* __restrict__ x, const float* __restrict__ gamma,
                 const unsigned short* __restrict__ wsq, float* __restrict__ outp)
{
    __shared__ __align__(16) unsigned short Qs[4][16 * C_DIM];  // 32 KB
    __shared__ __align__(16) unsigned short Ps[4][16 * 32];     //  4 KB

    const int lane = threadIdx.x & 31;
    const int wid  = threadIdx.x >> 5;
    const int ml   = lane & 15;
    const int hl   = lane >> 4;
    const int b    = blockIdx.y;
    const int i0   = (blockIdx.x * 4 + wid) * 16;
    const size_t NC = (size_t)N_DIM * C_DIM;

    const unsigned short* Qg = wsq;
    const unsigned short* Kt = wsq + 4 * NC;
    const unsigned short* Vm = wsq + 8 * NC;

    // Stage this wave's 16x256 bf16 Q tile into LDS (coalesced b128 copies).
    {
        const u32x4* src = (const u32x4*)(Qg + ((size_t)b * N_DIM + i0) * C_DIM);
        u32x4*       dst = (u32x4*)&Qs[wid][0];
        #pragma unroll
        for (int i = 0; i < 16; ++i) dst[lane + 32 * i] = src[lane + 32 * i];
    }
    asm volatile("s_wait_dscnt 0" ::: "memory");

    v8f O[16];
    {
        v8f z = {};
        #pragma unroll
        for (int f = 0; f < 16; ++f) O[f] = z;
    }
    float mrun[8], lrun[8];
    #pragma unroll
    for (int r = 0; r < 8; ++r) { mrun[r] = -INFINITY; lrun[r] = 0.0f; }

    for (int j0 = 0; j0 < N_DIM; j0 += 32) {
        // ---- S = Q * K^T for 2 column tiles of 16 ----
        v8f S0 = {}, S1 = {};
        const unsigned short* kr0 = Kt + ((size_t)b * N_DIM + j0 + ml) * C_DIM;
        const unsigned short* kr1 = kr0 + 16 * C_DIM;
        #pragma unroll
        for (int c0 = 0; c0 < C_DIM; c0 += 32) {
            Frag qa, kb;
            qa.q[0] = *(const u32x4*)&Qs[wid][ml * C_DIM + c0 + hl * 8];
            qa.q[1] = *(const u32x4*)&Qs[wid][ml * C_DIM + c0 + 16 + hl * 8];
            kb.q[0] = *(const u32x4*)(kr0 + c0 + hl * 8);
            kb.q[1] = *(const u32x4*)(kr0 + c0 + 16 + hl * 8);
            S0 = wmma_bf16(qa, kb, S0);
            kb.q[0] = *(const u32x4*)(kr1 + c0 + hl * 8);
            kb.q[1] = *(const u32x4*)(kr1 + c0 + 16 + hl * 8);
            S1 = wmma_bf16(qa, kb, S1);
        }
        if (j0 + 32 < N_DIM)
            __builtin_prefetch(kr0 + 32 * C_DIM, 0, 1);

        // ---- online softmax: row stats live replicated in each 16-lane half ----
        float alpha[8];
        bool  rescale = false;
        #pragma unroll
        for (int r = 0; r < 8; ++r) {
            float s = fmaxf(S0[r], S1[r]);
            s = fmaxf(s, __shfl_xor(s, 1, 32));
            s = fmaxf(s, __shfl_xor(s, 2, 32));
            s = fmaxf(s, __shfl_xor(s, 4, 32));
            s = fmaxf(s, __shfl_xor(s, 8, 32));
            const float mnew = fmaxf(mrun[r], s);
            rescale |= (mnew != mrun[r]);
            alpha[r] = __expf(mrun[r] - mnew);   // == 1.0f exactly if unchanged
            mrun[r]  = mnew;
            const float p0 = __expf(S0[r] - mnew);
            const float p1 = __expf(S1[r] - mnew);
            S0[r] = p0; S1[r] = p1;
            float rs = p0 + p1;
            rs += __shfl_xor(rs, 1, 32);
            rs += __shfl_xor(rs, 2, 32);
            rs += __shfl_xor(rs, 4, 32);
            rs += __shfl_xor(rs, 8, 32);
            lrun[r] = lrun[r] * alpha[r] + rs;
        }
        if (__any(rescale)) {      // wave-uniform skip of the 128-mul rescale
            #pragma unroll
            for (int f = 0; f < 16; ++f) {
                #pragma unroll
                for (int r = 0; r < 8; ++r) O[f][r] *= alpha[r];
            }
        }

        // ---- P: C-layout -> row-major LDS [16][32] -> A-fragment ----
        #pragma unroll
        for (int r = 0; r < 8; ++r) {
            const int row = r + 8 * hl;
            Ps[wid][row * 32 + ml]      = f2bf(S0[r]);
            Ps[wid][row * 32 + 16 + ml] = f2bf(S1[r]);
        }
        asm volatile("s_wait_dscnt 0" ::: "memory");
        Frag pa;
        pa.q[0] = *(const u32x4*)&Ps[wid][ml * 32 + hl * 8];
        pa.q[1] = *(const u32x4*)&Ps[wid][ml * 32 + 16 + hl * 8];

        // ---- O += P * V over all 256 channels ----
        #pragma unroll
        for (int f = 0; f < 16; ++f) {
            const int c0 = f * 16;
            const unsigned short* vr =
                Vm + ((size_t)b * C_DIM + c0 + ml) * N_DIM + j0;
            Frag vb;
            vb.q[0] = *(const u32x4*)(vr + hl * 8);
            vb.q[1] = *(const u32x4*)(vr + 16 + hl * 8);
            O[f] = wmma_bf16(pa, vb, O[f]);
        }
    }

    // ---- epilogue: out = gamma * O/l + x ----
    const float g = gamma[0];
    float inv[8];
    #pragma unroll
    for (int r = 0; r < 8; ++r) inv[r] = 1.0f / lrun[r];
    #pragma unroll
    for (int f = 0; f < 16; ++f) {
        const int c = f * 16 + ml;
        const size_t base = ((size_t)b * C_DIM + c) * N_DIM;
        #pragma unroll
        for (int r = 0; r < 8; ++r) {
            const int i = i0 + r + 8 * hl;
            outp[base + i] = g * (O[f][r] * inv[r]) + x[base + i];
        }
    }
}

// ---------------------------------------------------------------------------
extern "C" void kernel_launch(void* const* d_in, const int* in_sizes, int n_in,
                              void* d_out, int out_size, void* d_ws, size_t ws_size,
                              hipStream_t stream)
{
    (void)in_sizes; (void)n_in; (void)out_size; (void)ws_size;
    const float* x     = (const float*)d_in[0];
    const float* y     = (const float*)d_in[1];
    const float* wq    = (const float*)d_in[2];
    const float* wk    = (const float*)d_in[3];
    const float* wv    = (const float*)d_in[4];
    const float* gamma = (const float*)d_in[5];
    float*          outp = (float*)d_out;
    unsigned short* wsp  = (unsigned short*)d_ws;   // needs >= 24 MB
    const size_t NC = (size_t)N_DIM * C_DIM;

    // Q (x,wq) and Kt (y,wk): mode 0, blockIdx.y selects which; V (y,wv): mode 1.
    dim3 g0(1024, 2, B_DIM);
    proj_kernel<0><<<g0, 128, 0, stream>>>(x, wq, wsp,
                                           y, wk, wsp + 4 * NC);
    dim3 g1(1024, 1, B_DIM);
    proj_kernel<1><<<g1, 128, 0, stream>>>(y, wv, wsp + 8 * NC,
                                           y, wv, wsp + 8 * NC);

    // 256 query row-tiles per batch, 4 waves per block -> 64 blocks; y = batch
    dim3 g2(64, B_DIM, 1);
    attn_kernel<<<g2, 128, 0, stream>>>(x, gamma, wsp, outp);
}
